// PFAFM_79551384256641
// MI455X (gfx1250) — matrix operand order repaired and checked by reading, not compiled, not run on hardware
//
#include <hip/hip_runtime.h>
#include <math.h>

// ---------------------------------------------------------------------------
// PFAFM on gfx1250: all GEMM-shaped stages (convs, DFTs, IDFTs) run on bf16
// WMMA (v_wmma_f32_16x16x32_bf16, fp32 accum). fft2 / ifft2 are expressed as
// dense 128x128 DFT GEMMs (D symmetric); the 16384-pt IDFT over the flattened
// hw axis uses the four-step (128x128) decomposition with a twiddle epilogue.
// k_cgemm stages the block-shared A tile in LDS (8 waves share tm), padded to
// a 132-float row stride for conflict-free fragment reads.
// ---------------------------------------------------------------------------

typedef __attribute__((ext_vector_type(16))) __bf16 bf16x16;
typedef __attribute__((ext_vector_type(8)))  float  f32x8;

static constexpr int B_    = 8;
static constexpr int HW    = 16384;   // 128*128
static constexpr int WDIM  = 128;
static constexpr int LDA   = 132;     // padded LDS row stride (floats)

__device__ __forceinline__ f32x8 wmma_bf16(bf16x16 a, bf16x16 b, f32x8 c) {
  return __builtin_amdgcn_wmma_f32_16x16x32_bf16(false, a, false, b, (short)0, c,
                                                 false, false);
}

// Convert a gathered 16-float array into a bf16 fragment (optionally scaled).
__device__ __forceinline__ bf16x16 cvt_frag(const float* t, float sgn) {
  bf16x16 f;
#pragma unroll
  for (int i = 0; i < 16; ++i) f[i] = (__bf16)(sgn * t[i]);
  return f;
}

// A-fragment gather from LDS tile (16 rows x 128 cols, row stride LDA).
// The lane's 16 elements are row[k0+half*16 + 0..15] (matches frag_b128).
__device__ __forceinline__ void ld_a_lds(const float* lA, int k0, int lane,
                                         float* t) {
  int half = lane >> 4, r = lane & 15;
  const float* row = lA + r * LDA + k0 + half * 16;
#pragma unroll
  for (int i = 0; i < 8; ++i) { t[i] = row[i]; t[8 + i] = row[8 + i]; }
}

// B-fragment (32x16 slice of row-major K x N, ldb=128).
// lane<16: N=lane, K=k0+0..15 ; lane>=16: N=lane-16, K=k0+16..31.
__device__ __forceinline__ bf16x16 frag_b128(const float* __restrict__ base,
                                             int n0, int k0, int lane) {
  int half = lane >> 4, n = n0 + (lane & 15);
  const float* col = base + (size_t)(k0 + half * 16) * WDIM + n;
  float t[16];
#pragma unroll
  for (int j = 0; j < 16; ++j) t[j] = col[(size_t)j * WDIM];
  return cvt_frag(t, 1.0f);
}

// --------------------------- DFT tables ------------------------------------
// Dfwd[k*128+n] = exp(-2pi i k n /128) ; Dinv = exp(+2pi i k n /128)/128.
__global__ void k_init_dft(float* dfr, float* dfi, float* dir_, float* dii) {
  int idx = blockIdx.x * blockDim.x + threadIdx.x;
  if (idx >= WDIM * WDIM) return;
  int k = idx >> 7, n = idx & 127;
  int m = (k * n) & 127;                       // exact periodic reduction
  float th = 0.0490873852123405f * (float)m;   // 2*pi/128
  float s, c;
  __sincosf(th, &s, &c);
  dfr[idx] = c;  dfi[idx] = -s;
  dir_[idx] = c * (1.0f / 128.0f);
  dii[idx]  = s * (1.0f / 128.0f);
}

// --------------------- implicit-GEMM conv (+BN+ReLU) -----------------------
// NCHW, M = 16 pixels along W, N = 16 out channels, K = Cin per tap.
// Out-of-bounds taps: unconditional loads from clamped addresses, zeroed by a
// 0/1 multiplicative mask (no per-element EXEC predication).
__global__ void k_conv(const float* __restrict__ in, const float* __restrict__ w,
                       const float* __restrict__ cb, const float* __restrict__ gg,
                       const float* __restrict__ gb, const float* __restrict__ gm,
                       const float* __restrict__ gv, float* __restrict__ out,
                       int Cin, int Cout, int ks, int pad, int dil, int relu) {
  int lane = threadIdx.x & 31, wave = threadIdx.x >> 5;
  int tiles_n = Cout >> 4;
  int t = blockIdx.x * 8 + wave;
  int nt = t % tiles_n, mt = t / tiles_n;
  int wseg = mt & 7, y = (mt >> 3) & 127, b = mt >> 10;
  int half = lane >> 4, lr = lane & 15;
  int xm = wseg * 16 + lr;                     // pixel x for this lane's A row
  int oc = nt * 16 + lr;                       // out channel for B/C columns
  const float* inb = in + (size_t)b * Cin * HW;
  f32x8 acc = {};

  for (int ky = 0; ky < ks; ++ky) {
    int iy = y + ky * dil - pad;
    bool yok = (iy >= 0) && (iy < 128);
    int iyc = min(max(iy, 0), 127);
    for (int kx = 0; kx < ks; ++kx) {
      int ix = xm + kx * dil - pad;
      float msk = (yok && ix >= 0 && ix < 128) ? 1.0f : 0.0f;
      int ixc = min(max(ix, 0), 127);
      const float* src = inb + (size_t)iyc * WDIM + ixc;
      for (int k0 = 0; k0 < Cin; k0 += 32) {
        float ta[16], tb[16];
#pragma unroll
        for (int j = 0; j < 8; ++j) {
          int kc = k0 + 2 * j + half * 16;
          ta[2 * j]     = src[(size_t)kc * HW];
          ta[2 * j + 1] = src[(size_t)(kc + 1) * HW];
        }
        const float* wp = w + (((size_t)oc * Cin + k0 + half * 16) * ks + ky) * ks + kx;
#pragma unroll
        for (int j = 0; j < 16; ++j) tb[j] = wp[(size_t)j * ks * ks];
        acc = wmma_bf16(cvt_frag(ta, msk), cvt_frag(tb, 1.0f), acc);
      }
    }
  }
  float cbv = cb ? cb[oc] : 0.0f;
  float s = 1.0f, bias = 0.0f;
  if (gg) { s = gg[oc] * rsqrtf(gv[oc] + 1e-5f); bias = gb[oc] - gm[oc] * s; }
  float* dst = out + ((size_t)b * Cout + oc) * HW + (size_t)y * WDIM + wseg * 16;
#pragma unroll
  for (int r = 0; r < 8; ++r) {
    int m = r + 8 * half;
    float v = (acc[r] + cbv) * s + bias;
    if (relu) v = fmaxf(v, 0.0f);
    dst[m] = v;
  }
}

// --------------------- generic complex GEMM (128x128x128) ------------------
// C = scale * A x B per plane.  a_ps/b_ps == 0 -> shared (DFT matrix).
// a_im == nullptr -> real A.  epi: 0 none, 1 twiddle exp(+2pi i m n/16384),
// 2 abs (writes out_re only).  storeT: write C[m][n] to out[m + 128*n].
// All 8 waves of a block share tm -> the A tile (re+im) is staged once in LDS.
__global__ void k_cgemm(float* __restrict__ out_re, float* __restrict__ out_im,
                        const float* __restrict__ a_re, const float* __restrict__ a_im,
                        int a_ps,
                        const float* __restrict__ b_re, const float* __restrict__ b_im,
                        int b_ps, int storeT, int epi, float scale) {
  __shared__ float lA[2 * 16 * LDA];
  int lane = threadIdx.x & 31, wave = threadIdx.x >> 5;
  int t = blockIdx.x * 8 + wave;
  int tm = (t >> 3) & 7, tn = t & 7;       // tm == blockIdx.x & 7 (block-uniform)
  int p = blockIdx.y;
  const float* ar = a_re + (size_t)p * a_ps;
  const float* ai = a_im ? a_im + (size_t)p * a_ps : nullptr;
  const float* br = b_re + (size_t)p * b_ps;
  const float* bi = b_im + (size_t)p * b_ps;
  int m0 = tm * 16;

  // cooperative stage: 16x128 re (+ 16x128 im) -> LDS, padded rows
  {
    int tid = threadIdx.x;
#pragma unroll
    for (int q = 0; q < 2; ++q) {
      int f4 = tid * 2 + q;                // 0..511 float4 slots
      int row = f4 >> 5, c4 = (f4 & 31) * 4;
      float4 v = *(const float4*)(ar + (size_t)(m0 + row) * WDIM + c4);
      float* d = lA + row * LDA + c4;
      d[0] = v.x; d[1] = v.y; d[2] = v.z; d[3] = v.w;
    }
    if (ai) {
#pragma unroll
      for (int q = 0; q < 2; ++q) {
        int f4 = tid * 2 + q;
        int row = f4 >> 5, c4 = (f4 & 31) * 4;
        float4 v = *(const float4*)(ai + (size_t)(m0 + row) * WDIM + c4);
        float* d = lA + 16 * LDA + row * LDA + c4;
        d[0] = v.x; d[1] = v.y; d[2] = v.z; d[3] = v.w;
      }
    }
  }
  __syncthreads();

  f32x8 cre = {}, cim = {};
  for (int k0 = 0; k0 < 128; k0 += 32) {
    if (k0 < 96) {                        // prefetch next K-slice of B plane
      __builtin_prefetch(br + (size_t)(k0 + 32) * WDIM, 0, 1);
      __builtin_prefetch(bi + (size_t)(k0 + 32) * WDIM, 0, 1);
    }
    bf16x16 Bre = frag_b128(br, tn * 16, k0, lane);
    bf16x16 Bim = frag_b128(bi, tn * 16, k0, lane);
    float ta[16];
    ld_a_lds(lA, k0, lane, ta);
    bf16x16 Are = cvt_frag(ta, 1.0f);
    cre = wmma_bf16(Are, Bre, cre);
    cim = wmma_bf16(Are, Bim, cim);
    if (ai) {
      float ti[16];
      ld_a_lds(lA + 16 * LDA, k0, lane, ti);
      bf16x16 Aim  = cvt_frag(ti, 1.0f);
      bf16x16 AimN = cvt_frag(ti, -1.0f);
      cim = wmma_bf16(Aim, Bre, cim);
      cre = wmma_bf16(AimN, Bim, cre);   // Cre -= Aim*Bim
    }
  }
  int n  = tn * 16 + (lane & 15);
  int mb = tm * 16 + 8 * (lane >> 4);
  float* ore = out_re + (size_t)p * HW;
  float* oim = out_im ? out_im + (size_t)p * HW : nullptr;
#pragma unroll
  for (int r = 0; r < 8; ++r) {
    int m = mb + r;
    float re = cre[r] * scale, im = cim[r] * scale;
    if (epi == 1) {                       // four-step twiddle, inverse sign
      float ang = 3.8349519697141029e-4f * (float)(m * n);   // 2*pi/16384
      float s, c;
      __sincosf(ang, &s, &c);
      float r2 = re * c - im * s;
      im = re * s + im * c;
      re = r2;
    }
    size_t idx = storeT ? (size_t)(m + WDIM * n) : (size_t)m * WDIM + n;
    if (epi == 2) ore[idx] = sqrtf(re * re + im * im);
    else { ore[idx] = re; if (oim) oim[idx] = im; }
  }
}

// --------------------- fused 1x1 conv over concatenated parts --------------
__global__ void k_cat1x1(const float* p0, const float* p1, const float* p2,
                         const float* p3, const float* p4, int nparts,
                         int bcast_last, const float* __restrict__ w,
                         const float* cb, const float* gg, const float* gb,
                         const float* gm, const float* gv,
                         const float* __restrict__ res, float* __restrict__ out,
                         int Cout, int relu) {
  const float* parts[5] = {p0, p1, p2, p3, p4};
  int lane = threadIdx.x & 31, wave = threadIdx.x >> 5;
  int tiles_n = Cout >> 4;
  int t = blockIdx.x * 8 + wave;
  int nt = t % tiles_n, mt = t / tiles_n;
  int pix16 = mt & 1023, b = mt >> 10;
  int half = lane >> 4, lr = lane & 15;
  int pix = pix16 * 16 + lr;
  int oc  = nt * 16 + lr;
  int K = nparts * 32;
  f32x8 acc = {};
  for (int part = 0; part < nparts; ++part) {
    int k0 = part * 32;
    const float* P = parts[part];
    bool bc = bcast_last && (part == nparts - 1);
    float ta[16], tb[16];
    if (bc) {
#pragma unroll
      for (int j = 0; j < 8; ++j) {
        int pc = 2 * j + half * 16;
        ta[2 * j]     = P[b * 32 + pc];
        ta[2 * j + 1] = P[b * 32 + pc + 1];
      }
    } else {
#pragma unroll
      for (int j = 0; j < 8; ++j) {
        int pc = 2 * j + half * 16;
        ta[2 * j]     = P[((size_t)b * 32 + pc) * HW + pix];
        ta[2 * j + 1] = P[((size_t)b * 32 + pc + 1) * HW + pix];
      }
    }
    const float* wp = w + (size_t)oc * K + k0 + half * 16;
#pragma unroll
    for (int j = 0; j < 16; ++j) tb[j] = wp[j];
    acc = wmma_bf16(cvt_frag(ta, 1.0f), cvt_frag(tb, 1.0f), acc);
  }
  float cbv = cb ? cb[oc] : 0.0f;
  float s = 1.0f, bias = 0.0f;
  if (gg) { s = gg[oc] * rsqrtf(gv[oc] + 1e-5f); bias = gb[oc] - gm[oc] * s; }
#pragma unroll
  for (int r = 0; r < 8; ++r) {
    int m = r + 8 * half;
    size_t oidx = ((size_t)b * Cout + oc) * HW + pix16 * 16 + m;
    float v = (acc[r] + cbv) * s + bias;
    if (relu) v = fmaxf(v, 0.0f);
    if (res) v += res[oidx];
    out[oidx] = v;
  }
}

// --------------------------- reductions ------------------------------------
__global__ void k_gap(const float* __restrict__ x, float* __restrict__ gap) {
  int bc = blockIdx.x;
  const float* s = x + (size_t)bc * HW;
  float acc = 0.0f;
  for (int i = threadIdx.x; i < HW; i += 256) acc += s[i];
  __shared__ float red[256];
  red[threadIdx.x] = acc;
  __syncthreads();
  for (int st = 128; st > 0; st >>= 1) {
    if (threadIdx.x < st) red[threadIdx.x] += red[threadIdx.x + st];
    __syncthreads();
  }
  if (threadIdx.x == 0) gap[bc] = red[0] * (1.0f / HW);
}

__global__ void k_c5(const float* __restrict__ gap, const float* __restrict__ w,
                     const float* cb, const float* gg, const float* gb,
                     const float* gm, const float* gv, float* __restrict__ c5v) {
  int tid = threadIdx.x;            // 256 = 8 batches x 32 channels
  int b = tid >> 5, oc = tid & 31;
  float acc = cb[oc];
  for (int ic = 0; ic < 64; ++ic) acc += gap[b * 64 + ic] * w[oc * 64 + ic];
  float s = gg[oc] * rsqrtf(gv[oc] + 1e-5f);
  float v = acc * s + (gb[oc] - gm[oc] * s);
  c5v[b * 32 + oc] = fmaxf(v, 0.0f);
}

__global__ void k_norms(const float* __restrict__ fr, const float* __restrict__ fi,
                        float* __restrict__ nrm) {
  int row = blockIdx.x;             // b*32 + ch, 256 rows
  const float* r = fr + (size_t)row * HW;
  const float* i = fi + (size_t)row * HW;
  float acc = 0.0f;
  for (int n = threadIdx.x; n < HW; n += 256) acc += r[n] * r[n] + i[n] * i[n];
  __shared__ float red[256];
  red[threadIdx.x] = acc;
  __syncthreads();
  for (int st = 128; st > 0; st >>= 1) {
    if (threadIdx.x < st) red[threadIdx.x] += red[threadIdx.x + st];
    __syncthreads();
  }
  if (threadIdx.x == 0) nrm[row] = fmaxf(sqrtf(red[0]), 1e-12f);
}

// ------------------- attention: S = qk.qk^T, dual softmax -------------------
__global__ void k_attn(const float* __restrict__ fr, const float* __restrict__ fi,
                       const float* __restrict__ nrm, const float* __restrict__ temp,
                       float* __restrict__ are, float* __restrict__ aim) {
  int bh = blockIdx.x, b = bh >> 3, h = bh & 7;
  const float* vr = fr + ((size_t)b * 32 + h * 4) * HW;
  const float* vi = fi + ((size_t)b * 32 + h * 4) * HW;
  float sr[16] = {}, si[16] = {};
  for (int n = threadIdx.x; n < HW; n += 256) {
    float xr[4], xi[4];
#pragma unroll
    for (int c = 0; c < 4; ++c) { xr[c] = vr[c * HW + n]; xi[c] = vi[c * HW + n]; }
#pragma unroll
    for (int c = 0; c < 4; ++c)
#pragma unroll
      for (int d = 0; d < 4; ++d) {
        sr[c * 4 + d] += xr[c] * xr[d] - xi[c] * xi[d];
        si[c * 4 + d] += xr[c] * xi[d] + xi[c] * xr[d];
      }
  }
  __shared__ float red[256];
  __shared__ float fin[32];
  for (int comp = 0; comp < 32; ++comp) {
    red[threadIdx.x] = comp < 16 ? sr[comp] : si[comp - 16];
    __syncthreads();
    for (int st = 128; st > 0; st >>= 1) {
      if (threadIdx.x < st) red[threadIdx.x] += red[threadIdx.x + st];
      __syncthreads();
    }
    if (threadIdx.x == 0) fin[comp] = red[0];
    __syncthreads();
  }
  if (threadIdx.x == 0) {
    float T = temp[h];
    float R[16], I[16];
    for (int c = 0; c < 4; ++c)
      for (int d = 0; d < 4; ++d) {
        float nn = nrm[b * 32 + h * 4 + c] * nrm[b * 32 + h * 4 + d];
        R[c * 4 + d] = fin[c * 4 + d] / nn * T;
        I[c * 4 + d] = fin[16 + c * 4 + d] / nn * T;
      }
    for (int c = 0; c < 4; ++c) {     // softmax along d, real and imag
      float mr = R[c * 4], mi = I[c * 4];
      for (int d = 1; d < 4; ++d) { mr = fmaxf(mr, R[c * 4 + d]); mi = fmaxf(mi, I[c * 4 + d]); }
      float srx = 0.0f, six = 0.0f;
      for (int d = 0; d < 4; ++d) {
        R[c * 4 + d] = __expf(R[c * 4 + d] - mr); srx += R[c * 4 + d];
        I[c * 4 + d] = __expf(I[c * 4 + d] - mi); six += I[c * 4 + d];
      }
      for (int d = 0; d < 4; ++d) { R[c * 4 + d] /= srx; I[c * 4 + d] /= six; }
    }
    for (int i = 0; i < 16; ++i) { are[bh * 16 + i] = R[i]; aim[bh * 16 + i] = I[i]; }
  }
}

// o[c,n] = sum_d attn[c,d] * v[d,n]   (complex, per-pixel)
__global__ void k_apply(const float* __restrict__ fr, const float* __restrict__ fi,
                        const float* __restrict__ are, const float* __restrict__ aim,
                        float* __restrict__ orr, float* __restrict__ oii) {
  size_t idx = (size_t)blockIdx.x * 256 + threadIdx.x;   // B*8*HW
  int pix = idx & (HW - 1);
  int bh = idx >> 14, b = bh >> 3, h = bh & 7;
  size_t base = ((size_t)b * 32 + h * 4) * HW + pix;
  float vr[4], vi[4];
#pragma unroll
  for (int d = 0; d < 4; ++d) { vr[d] = fr[base + d * HW]; vi[d] = fi[base + d * HW]; }
#pragma unroll
  for (int c = 0; c < 4; ++c) {
    float orx = 0.0f, oix = 0.0f;
#pragma unroll
    for (int d = 0; d < 4; ++d) {
      float arr = are[bh * 16 + c * 4 + d], aii = aim[bh * 16 + c * 4 + d];
      orx += arr * vr[d] - aii * vi[d];
      oix += arr * vi[d] + aii * vr[d];
    }
    orr[base + c * HW] = orx;
    oii[base + c * HW] = oix;
  }
}

// 4-point IDFT across head-channels + abs  ->  out_f
__global__ void k_ifft4_abs(const float* __restrict__ rr, const float* __restrict__ ri,
                            float* __restrict__ outf) {
  size_t idx = (size_t)blockIdx.x * 256 + threadIdx.x;   // B*8*HW
  int pix = idx & (HW - 1);
  int bh = idx >> 14, b = bh >> 3, h = bh & 7;
  size_t base = ((size_t)b * 32 + h * 4) * HW + pix;
  float xr[4], xi[4];
#pragma unroll
  for (int c = 0; c < 4; ++c) { xr[c] = rr[base + c * HW]; xi[c] = ri[base + c * HW]; }
  const float phr[4] = {1.f, 0.f, -1.f, 0.f};
  const float phi[4] = {0.f, 1.f, 0.f, -1.f};
#pragma unroll
  for (int q = 0; q < 4; ++q) {
    float sr = 0.0f, si = 0.0f;
#pragma unroll
    for (int c = 0; c < 4; ++c) {
      int p = (c * q) & 3;
      sr += xr[c] * phr[p] - xi[c] * phi[p];
      si += xr[c] * phi[p] + xi[c] * phr[p];
    }
    outf[base + q * HW] = 0.25f * sqrtf(sr * sr + si * si);
  }
}

// gate stage 1: 1x1 conv 32->2 + BN + ReLU on f.real
__global__ void k_gate1(const float* __restrict__ fr, const float* __restrict__ w1,
                        const float* b1, const float* gg, const float* gb,
                        const float* gm, const float* gv, float* __restrict__ g1) {
  size_t idx = (size_t)blockIdx.x * 256 + threadIdx.x;   // B*HW
  int pix = idx & (HW - 1), b = idx >> 14;
#pragma unroll
  for (int gch = 0; gch < 2; ++gch) {
    float acc = b1[gch];
    for (int ic = 0; ic < 32; ++ic)
      acc += fr[((size_t)b * 32 + ic) * HW + pix] * w1[gch * 32 + ic];
    float s = gg[gch] * rsqrtf(gv[gch] + 1e-5f);
    float v = acc * s + (gb[gch] - gm[gch] * s);
    g1[((size_t)b * 2 + gch) * HW + pix] = fmaxf(v, 0.0f);
  }
}

// gate stage 2: 1x1 conv 2->32 + sigmoid, then gf = gate * f (complex)
__global__ void k_gate2mul(const float* __restrict__ g1, const float* __restrict__ w2,
                           const float* b2, const float* __restrict__ fr,
                           const float* __restrict__ fi, float* __restrict__ gr,
                           float* __restrict__ gi) {
  size_t idx = (size_t)blockIdx.x * 256 + threadIdx.x;   // B*HW
  int pix = idx & (HW - 1), b = idx >> 14;
  float g0 = g1[((size_t)b * 2 + 0) * HW + pix];
  float g1v = g1[((size_t)b * 2 + 1) * HW + pix];
  for (int oc = 0; oc < 32; ++oc) {
    float z = g0 * w2[oc * 2] + g1v * w2[oc * 2 + 1] + b2[oc];
    float gate = 1.0f / (1.0f + __expf(-z));
    size_t fidx = ((size_t)b * 32 + oc) * HW + pix;
    gr[fidx] = gate * fr[fidx];
    gi[fidx] = gate * fi[fidx];
  }
}

// final 1x1 conv 16->1 (+bias)
__global__ void k_out2(const float* __restrict__ o1, const float* __restrict__ w,
                       const float* bias, float* __restrict__ out) {
  size_t idx = (size_t)blockIdx.x * 256 + threadIdx.x;   // B*HW
  int pix = idx & (HW - 1), b = idx >> 14;
  float acc = bias[0];
#pragma unroll
  for (int ic = 0; ic < 16; ++ic)
    acc += o1[((size_t)b * 16 + ic) * HW + pix] * w[ic];
  out[idx] = acc;
}

// ---------------------------------------------------------------------------
extern "C" void kernel_launch(void* const* d_in, const int* in_sizes, int n_in,
                              void* d_out, int out_size, void* d_ws, size_t ws_size,
                              hipStream_t stream) {
  (void)in_sizes; (void)n_in; (void)out_size; (void)ws_size;
  auto F = [&](int i) { return (const float*)d_in[i]; };
  // JAX pytree (sorted-dict) flatten order:
  // params: c1(0) c2(6) c3(12) c4(18) c5(24) down(30) fuse(36)
  //         gate{b1,b2,bn.b,bn.g,bn.m,bn.v,w1,w2}(42..49)
  //         out1(50) out2{b,w}(56,57) proj(58) temp(59) ; x(60)
  // each cbrp block: {b, bn.b, bn.g, bn.m, bn.v, w}
  const int C1 = 0, C2 = 6, C3 = 12, C4 = 18, C5 = 24, DW = 30, FU = 36;
  const int GB1 = 42, GB2 = 43, GBNB = 44, GBNG = 45, GBNM = 46, GBNV = 47,
            GW1 = 48, GW2 = 49, O1 = 50, O2B = 56, O2W = 57, PRJ = 58, TMP = 59,
            XI = 60;

  float* W = (float*)d_ws;
  size_t off = 0;
  auto alloc = [&](size_t n) { float* p = W + off; off += n; return p; };
  const size_t P32 = (size_t)B_ * 32 * HW;
  float* xd    = alloc((size_t)B_ * 64 * HW);
  float* c1b   = alloc(P32);
  float* Fb0   = alloc(P32);
  float* Fb1   = alloc(P32);
  float* Fb2   = alloc(P32);
  float* convb = alloc(P32);
  float* Are   = alloc(P32);
  float* Aim   = alloc(P32);
  float* Bre   = alloc(P32);
  float* Bim   = alloc(P32);
  float* Cre   = alloc(P32);
  float* Cim   = alloc(P32);
  float* outf  = alloc(P32);
  float* outl  = alloc(P32);
  float* g1b   = alloc((size_t)B_ * 2 * HW);
  float* dfr   = alloc(HW);
  float* dfi   = alloc(HW);
  float* dir_  = alloc(HW);
  float* dii   = alloc(HW);
  float* gap   = alloc(512);
  float* c5v   = alloc(256);
  float* nrm   = alloc(256);
  float* atr   = alloc(1024);
  float* ati   = alloc(1024);
  float* Fbs[3] = {Fb0, Fb1, Fb2};

  k_init_dft<<<64, 256, 0, stream>>>(dfr, dfi, dir_, dii);

  // down: 3x3 64->64 +BN+ReLU
  k_conv<<<4096, 256, 0, stream>>>(F(XI), F(DW + 5), F(DW), F(DW + 2), F(DW + 1),
                                   F(DW + 3), F(DW + 4), xd, 64, 64, 3, 1, 1, 1);
  k_gap<<<512, 256, 0, stream>>>(xd, gap);
  k_c5<<<1, 256, 0, stream>>>(gap, F(C5 + 5), F(C5), F(C5 + 2), F(C5 + 1),
                              F(C5 + 3), F(C5 + 4), c5v);
  // c1: 1x1 64->32
  k_conv<<<2048, 256, 0, stream>>>(xd, F(C1 + 5), F(C1), F(C1 + 2), F(C1 + 1),
                                   F(C1 + 3), F(C1 + 4), c1b, 64, 32, 1, 0, 1, 1);

  const int cbase[3] = {C2, C3, C4};
  const int dils[3] = {2, 4, 6};
  dim3 ggrid(8, 256);
  for (int i = 0; i < 3; ++i) {
    int cb = cbase[i], dl = dils[i];
    // dilated 3x3 64->32
    k_conv<<<2048, 256, 0, stream>>>(xd, F(cb + 5), F(cb), F(cb + 2), F(cb + 1),
                                     F(cb + 3), F(cb + 4), convb, 64, 32, 3, dl, dl, 1);
    // fft2: Y = X @ Dfwd (sym)   then  F = Dfwd @ Y
    k_cgemm<<<ggrid, 256, 0, stream>>>(Are, Aim, convb, nullptr, HW, dfr, dfi, 0,
                                       0, 0, 1.0f);
    k_cgemm<<<ggrid, 256, 0, stream>>>(Bre, Bim, dfr, dfi, 0, Are, Aim, HW,
                                       0, 0, 1.0f);
    // attention
    k_norms<<<256, 256, 0, stream>>>(Bre, Bim, nrm);
    k_attn<<<64, 256, 0, stream>>>(Bre, Bim, nrm, F(TMP), atr, ati);
    k_apply<<<4096, 256, 0, stream>>>(Bre, Bim, atr, ati, Are, Aim);
    // 16384-pt IDFT (four-step): col IDFT + twiddle, row IDFT + transpose store
    k_cgemm<<<ggrid, 256, 0, stream>>>(Cre, Cim, dir_, dii, 0, Are, Aim, HW,
                                       0, 1, 1.0f);
    k_cgemm<<<ggrid, 256, 0, stream>>>(Are, Aim, Cre, Cim, HW, dir_, dii, 0,
                                       1, 0, 1.0f);
    k_ifft4_abs<<<4096, 256, 0, stream>>>(Are, Aim, outf);
    // gate path: gf = sigmoid(conv(relu(bn(conv(f.real))))) * f ; abs(ifft2(gf))
    k_gate1<<<512, 256, 0, stream>>>(Bre, F(GW1), F(GB1), F(GBNG), F(GBNB),
                                     F(GBNM), F(GBNV), g1b);
    k_gate2mul<<<512, 256, 0, stream>>>(g1b, F(GW2), F(GB2), Bre, Bim, Cre, Cim);
    k_cgemm<<<ggrid, 256, 0, stream>>>(Are, Aim, Cre, Cim, HW, dir_, dii, 0,
                                       0, 0, 1.0f);
    k_cgemm<<<ggrid, 256, 0, stream>>>(outl, nullptr, dir_, dii, 0, Are, Aim, HW,
                                       0, 2, 1.0f);
    // proj(concat(out_f,out_l)) + conv_out
    k_cat1x1<<<2048, 256, 0, stream>>>(outf, outl, nullptr, nullptr, nullptr, 2, 0,
                                       F(PRJ), nullptr, nullptr, nullptr, nullptr,
                                       nullptr, convb, Fbs[i], 32, 0);
  }

  // fuse: 1x1 160->32 over [c1, F2, F3, F4, broadcast(c5)]
  float* fused = Are;
  k_cat1x1<<<2048, 256, 0, stream>>>(c1b, Fb0, Fb1, Fb2, c5v, 5, 1, F(FU + 5),
                                     F(FU), F(FU + 2), F(FU + 1), F(FU + 3),
                                     F(FU + 4), nullptr, fused, 32, 1);
  // out1: 3x3 32->16
  float* o1b = Bre;
  k_conv<<<1024, 256, 0, stream>>>(fused, F(O1 + 5), F(O1), F(O1 + 2), F(O1 + 1),
                                   F(O1 + 3), F(O1 + 4), o1b, 32, 16, 3, 1, 1, 1);
  // out2: 1x1 16->1
  k_out2<<<512, 256, 0, stream>>>(o1b, F(O2W), F(O2B), (float*)d_out);
}
